// MultiHeadAttention_v2_79405355369264
// MI455X (gfx1250) — matrix operand order repaired and checked by reading, not compiled, not run on hardware
//
#include <hip/hip_runtime.h>

// ---------------------------------------------------------------------------
// MHA (B=4, S=2048, D=2048, H=16, HD=128) for gfx1250 (MI455X).
// bf16 WMMA (v_wmma_f32_16x16x32_bf16) + async global->LDS double buffering.
// ---------------------------------------------------------------------------

typedef __bf16 v16bf __attribute__((ext_vector_type(16)));
typedef float  v8f   __attribute__((ext_vector_type(8)));
typedef unsigned short u16;
typedef unsigned int   u32;
typedef unsigned long long u64;

#define BB  4
#define SS  2048
#define DD  2048
#define HH  16
#define HDD 128

// float -> bf16 (round-to-nearest-even), bf16 -> float, via bit ops only.
__device__ __forceinline__ u16 f2bf(float f) {
  u32 u = __float_as_uint(f);
  u32 r = u + 0x7FFFu + ((u >> 16) & 1u);
  return (u16)(r >> 16);
}
__device__ __forceinline__ float bf2f(u16 s) {
  return __uint_as_float(((u32)s) << 16);
}
__device__ __forceinline__ u16 to_bf(float f) { return f2bf(f); }
__device__ __forceinline__ u16 to_bf(u16 s)   { return s; }

__device__ __forceinline__ v8f wmma_bf16(v16bf a, v16bf b, v8f c) {
  return __builtin_amdgcn_wmma_f32_16x16x32_bf16(false, a, false, b,
                                                 (short)0, c, false, false);
}

// LDS byte offset of a shared-memory object: low 32 bits of the generic
// address (aperture lives in bits 63:32 per the CDNA5 aperture check).
__device__ __forceinline__ u32 lds_off(const void* p) {
  return (u32)(uintptr_t)p;
}
// Async 16B global -> LDS copy, per lane.  Tracked by ASYNCcnt.
__device__ __forceinline__ void async_b128(u32 lds_byte, const void* gptr) {
  asm volatile("global_load_async_to_lds_b128 %0, %1, off"
               :: "v"(lds_byte), "v"((u64)(uintptr_t)gptr) : "memory");
}
__device__ __forceinline__ void wait_async0() {
  asm volatile("s_wait_asynccnt 0" ::: "memory");
}

// ---------------------------------------------------------------------------
// fp32 -> bf16 cast, 4 elements / thread (for x)
// ---------------------------------------------------------------------------
__global__ __launch_bounds__(256) void cast_bf16_kernel(
    const float* __restrict__ src, u16* __restrict__ dst, int n4) {
  int i = blockIdx.x * blockDim.x + threadIdx.x;
  if (i < n4) {
    float4 v = ((const float4*)src)[i];
    u16 o[4] = {f2bf(v.x), f2bf(v.y), f2bf(v.z), f2bf(v.w)};
    *(uint2*)&dst[(size_t)i * 4] = *(uint2*)o;
  }
}

// ---------------------------------------------------------------------------
// Tiled transpose (+ optional fp32->bf16): src [R][C] -> dst [C][R], batched.
// 64x64 tiles through LDS; coalesced on both sides.
// ---------------------------------------------------------------------------
template <typename SRC>
__global__ __launch_bounds__(256) void transpose_bf16_kernel(
    const SRC* __restrict__ src, u16* __restrict__ dst, int R, int C,
    size_t sbatch, size_t dbatch) {
  __shared__ u16 L[64][72];
  const int t  = threadIdx.x;
  const int r0 = blockIdx.y * 64;
  const int c0 = blockIdx.x * 64;
  src += (size_t)blockIdx.z * sbatch;
  dst += (size_t)blockIdx.z * dbatch;
  {
    const int lr = t >> 2, lc = (t & 3) * 16;
    const SRC* sp = src + (size_t)(r0 + lr) * C + c0 + lc;
#pragma unroll
    for (int j = 0; j < 16; ++j) L[lr][lc + j] = to_bf(sp[j]);
  }
  __syncthreads();
  {
    const int dr = t >> 2, dc = (t & 3) * 16;
    u16 tmp[16];
#pragma unroll
    for (int j = 0; j < 16; ++j) tmp[j] = L[dc + j][dr];
    u16* dp = dst + (size_t)(c0 + dr) * R + r0 + dc;
    *(uint4*)&dp[0] = *(uint4*)&tmp[0];
    *(uint4*)&dp[8] = *(uint4*)&tmp[8];
  }
}

// ---------------------------------------------------------------------------
// bf16 GEMM: C[M,N] = A[M,K] * Bt[N,K]^T.  Block tile 64x128, BK=32, 8 waves,
// each wave 32x32 (2x2 WMMA frags).  Double-buffered async global->LDS.
// MODE 0: fp32 C + bias.  MODE 1: scatter bf16 q/k/v into [B,H,S,HD].
// ---------------------------------------------------------------------------
template <int MODE>
__global__ __launch_bounds__(256) void gemm_bf16_kernel(
    const u16* __restrict__ A, const u16* __restrict__ Bt,
    const float* __restrict__ bias, float* __restrict__ Cf,
    u16* __restrict__ qout, u16* __restrict__ kout, u16* __restrict__ vout,
    int M, int N, int K) {
  __shared__ u16 As[2][64 * 40];    // [m][k], rows padded to 40
  __shared__ u16 Bs[2][128 * 40];   // [n][k], rows padded to 40

  const int tid  = threadIdx.x;
  const int w    = tid >> 5;
  const int lane = tid & 31;
  const int ml   = lane & 15;
  const int half = lane >> 4;
  const int wm   = w >> 2;          // 0..1
  const int wn   = w & 3;           // 0..3
  const int m0   = blockIdx.y * 64;
  const int n0   = blockIdx.x * 128;

  const int arow = tid >> 2;        // 0..63
  const int acol = (tid & 3) * 8;   // 0,8,16,24
  const int brow = tid >> 1;        // 0..127
  const int bcol = (tid & 1) * 16;  // 0,16

  auto stage = [&](int buf, int k0) {
    async_b128(lds_off(&As[buf][arow * 40 + acol]),
               A + (size_t)(m0 + arow) * K + k0 + acol);
    const u16* bp = Bt + (size_t)(n0 + brow) * K + k0 + bcol;
    const u32 bl = lds_off(&Bs[buf][brow * 40 + bcol]);
    async_b128(bl,      bp);
    async_b128(bl + 16, bp + 8);
  };

  v8f acc[2][2];
#pragma unroll
  for (int i = 0; i < 2; ++i)
#pragma unroll
    for (int j = 0; j < 2; ++j)
#pragma unroll
      for (int r = 0; r < 8; ++r) acc[i][j][r] = 0.f;

  const int iters = K / 32;
  stage(0, 0);
  for (int it = 0; it < iters; ++it) {
    wait_async0();       // my async loads for buf[it&1] have landed
    __syncthreads();     // everyone's have; prev compute on buf[1-cur] done
    if (it + 1 < iters) stage((it + 1) & 1, (it + 1) * 32);
    const int cur = it & 1;

    v16bf af[2], bfm[2];
#pragma unroll
    for (int i = 0; i < 2; ++i) {
      const u16* p = &As[cur][(wm * 32 + i * 16 + ml) * 40 + half * 8];
      ((uint4*)&af[i])[0] = *(const uint4*)p;
      ((uint4*)&af[i])[1] = *(const uint4*)(p + 16);
    }
#pragma unroll
    for (int j = 0; j < 2; ++j) {
      const u16* p = &Bs[cur][(wn * 32 + j * 16 + ml) * 40 + half * 8];
      ((uint4*)&bfm[j])[0] = *(const uint4*)p;
      ((uint4*)&bfm[j])[1] = *(const uint4*)(p + 16);
    }
#pragma unroll
    for (int i = 0; i < 2; ++i)
#pragma unroll
      for (int j = 0; j < 2; ++j)
        acc[i][j] = wmma_bf16(af[i], bfm[j], acc[i][j]);
  }

  // --- epilogue ---
#pragma unroll
  for (int i = 0; i < 2; ++i) {
    const int mtop = m0 + wm * 32 + i * 16 + 8 * half;  // + r
#pragma unroll
    for (int j = 0; j < 2; ++j) {
      const int ncol0 = n0 + wn * 32 + j * 16;
      const int n = ncol0 + ml;
      if constexpr (MODE == 0) {
        const float bv = bias[n];
#pragma unroll
        for (int r = 0; r < 8; ++r)
          Cf[(size_t)(mtop + r) * N + n] = acc[i][j][r] + bv;
      } else {
        const int which = ncol0 >> 11;  // 0=q 1=k 2=v (tile never straddles)
        u16* dst = (which == 0) ? qout : ((which == 1) ? kout : vout);
        const int d  = n & (DD - 1);
        const int hh = d >> 7;
        const int hd = d & (HDD - 1);
#pragma unroll
        for (int r = 0; r < 8; ++r) {
          const int m = mtop + r;
          const int b = m >> 11;        // row = b*S + s
          const int s = m & (SS - 1);
          dst[(((size_t)(b * HH + hh)) * SS + s) * HDD + hd] = f2bf(acc[i][j][r]);
        }
      }
    }
  }
}

// ---------------------------------------------------------------------------
// NeoX RoPE, in-place on bf16 q and k ([B*H, S, HD]); one thread per pair.
// ---------------------------------------------------------------------------
__global__ __launch_bounds__(256) void rope_kernel(u16* __restrict__ q,
                                                   u16* __restrict__ k) {
  const int idx = blockIdx.x * blockDim.x + threadIdx.x;
  const int total = BB * HH * SS * (HDD / 2);
  if (idx >= total) return;
  const int i  = idx & 63;
  const int s  = (idx >> 6) & (SS - 1);
  const int bh = idx >> 17;
  const size_t base = ((size_t)bh * SS + s) * HDD;

  const float theta = __powf(10000.f, -(float)i * (1.0f / 64.0f));
  const float ang = (float)s * theta;
  const float c = cosf(ang), sn = sinf(ang);

  float q0 = bf2f(q[base + i]), q1 = bf2f(q[base + i + 64]);
  q[base + i]      = f2bf(q0 * c - q1 * sn);
  q[base + i + 64] = f2bf(q1 * c + q0 * sn);

  float k0 = bf2f(k[base + i]), k1 = bf2f(k[base + i + 64]);
  k[base + i]      = f2bf(k0 * c - k1 * sn);
  k[base + i + 64] = f2bf(k1 * c + k0 * sn);
}

// ---------------------------------------------------------------------------
// Flash attention: block = 4 waves, 64 queries; each wave owns 16 q-rows.
// K in [B,H,S,HD]; V pre-transposed to [B,H,HD,S].  32-key tiles, double-
// buffered async staging, causal tiles skipped wave-uniformly.
// ---------------------------------------------------------------------------
__global__ __launch_bounds__(128) void attn_kernel(
    const u16* __restrict__ qbf, const u16* __restrict__ kbf,
    const u16* __restrict__ vtbf, u16* __restrict__ obf) {
  __shared__ u16 Ks[2][32 * 136];   // [key][hd], rows padded to 136
  __shared__ u16 Vs[2][128 * 40];   // [hd][key], rows padded to 40
  __shared__ u16 Ps[4 * 16 * 40];   // per-wave P scratch [16][32], padded

  const int tid  = threadIdx.x;
  const int w    = tid >> 5;
  const int lane = tid & 31;
  const int ml   = lane & 15;
  const int half = lane >> 4;

  const int qt = blockIdx.x & 31;               // S/64 = 32
  const int h  = (blockIdx.x >> 5) & (HH - 1);
  const int b  = blockIdx.x >> 9;
  const int q0 = qt * 64;
  const size_t headbase = ((size_t)(b * HH + h)) * SS * HDD;
  const u16* qh  = qbf  + headbase;
  const u16* kh  = kbf  + headbase;
  const u16* vth = vtbf + headbase;             // [HD][S]

  // Q fragments for this wave's 16 rows (A-layout, 4 k-steps over HD=128).
  v16bf qa[4];
  {
    const int qrow = q0 + w * 16 + ml;
#pragma unroll
    for (int kc = 0; kc < 4; ++kc) {
      const u16* p = qh + (size_t)qrow * HDD + kc * 32 + half * 8;
      ((uint4*)&qa[kc])[0] = *(const uint4*)p;
      ((uint4*)&qa[kc])[1] = *(const uint4*)(p + 16);
    }
  }

  float mi[8], li[8];
  v8f O[8];
#pragma unroll
  for (int r = 0; r < 8; ++r) { mi[r] = -3.0e38f; li[r] = 0.f; }
#pragma unroll
  for (int nj = 0; nj < 8; ++nj)
#pragma unroll
    for (int r = 0; r < 8; ++r) O[nj][r] = 0.f;

  const int qbase_w = q0 + w * 16;
  const float scale = 0.08838834764831845f;     // 1/sqrt(128)

  const int kl_key = tid >> 2;                  // 0..31
  const int kl_col = (tid & 3) * 32;            // 0,32,64,96

  auto stage = [&](int buf, int j0) {
    const u16* ks = kh + (size_t)(j0 + kl_key) * HDD + kl_col;
    const u32 kd = lds_off(&Ks[buf][kl_key * 136 + kl_col]);
#pragma unroll
    for (int qq = 0; qq < 4; ++qq) async_b128(kd + qq * 16, ks + qq * 8);
    const u16* vs = vth + (size_t)tid * SS + j0;  // 32 keys for hd=tid
    const u32 vd = lds_off(&Vs[buf][tid * 40]);
#pragma unroll
    for (int qq = 0; qq < 4; ++qq) async_b128(vd + qq * 16, vs + qq * 8);
  };

  const int ntiles = (q0 + 64) / 32;
  stage(0, 0);
  for (int it = 0; it < ntiles; ++it) {
    wait_async0();
    __syncthreads();
    if (it + 1 < ntiles) stage((it + 1) & 1, (it + 1) * 32);
    const int cur = it & 1;
    const int j0 = it * 32;
    if (j0 > qbase_w + 15) continue;  // tile fully above diagonal for wave

    // --- S = Q K^T for two 16-key groups ---
    v8f s0, s1;
#pragma unroll
    for (int r = 0; r < 8; ++r) { s0[r] = 0.f; s1[r] = 0.f; }
#pragma unroll
    for (int kc = 0; kc < 4; ++kc) {
      v16bf b0, b1;
      const u16* p0 = &Ks[cur][ml * 136 + kc * 32 + half * 8];
      const u16* p1 = &Ks[cur][(16 + ml) * 136 + kc * 32 + half * 8];
      ((uint4*)&b0)[0] = *(const uint4*)p0;
      ((uint4*)&b0)[1] = *(const uint4*)(p0 + 16);
      ((uint4*)&b1)[0] = *(const uint4*)p1;
      ((uint4*)&b1)[1] = *(const uint4*)(p1 + 16);
      s0 = wmma_bf16(qa[kc], b0, s0);
      s1 = wmma_bf16(qa[kc], b1, s1);
    }

    // --- scale, causal mask, online softmax ---
    const int key0 = j0 + ml;
    const int key1 = key0 + 16;
    const bool need_mask = (j0 + 31 > qbase_w);
#pragma unroll
    for (int r = 0; r < 8; ++r) {
      const int qi = qbase_w + r + 8 * half;
      float a0 = s0[r] * scale;
      float a1 = s1[r] * scale;
      if (need_mask) {
        if (key0 > qi) a0 = -3.0e38f;
        if (key1 > qi) a1 = -3.0e38f;
      }
      float t = fmaxf(a0, a1);
      t = fmaxf(t, __shfl_xor(t, 1));
      t = fmaxf(t, __shfl_xor(t, 2));
      t = fmaxf(t, __shfl_xor(t, 4));
      t = fmaxf(t, __shfl_xor(t, 8));
      const float mnew  = fmaxf(mi[r], t);
      const float alpha = __expf(mi[r] - mnew);
      mi[r] = mnew;
      const float p0 = __expf(a0 - mnew);
      const float p1 = __expf(a1 - mnew);
      float rs = p0 + p1;
      rs += __shfl_xor(rs, 1);
      rs += __shfl_xor(rs, 2);
      rs += __shfl_xor(rs, 4);
      rs += __shfl_xor(rs, 8);
      li[r] = li[r] * alpha + rs;
#pragma unroll
      for (int nj = 0; nj < 8; ++nj) O[nj][r] *= alpha;
      Ps[(w * 16 + r + 8 * half) * 40 + ml]      = f2bf(p0);
      Ps[(w * 16 + r + 8 * half) * 40 + 16 + ml] = f2bf(p1);
    }

    // --- reload P as an A fragment (16x32), then O += P V ---
    v16bf pa;
    {
      const u16* pp = &Ps[(w * 16 + ml) * 40 + half * 8];
      ((uint4*)&pa)[0] = *(const uint4*)pp;
      ((uint4*)&pa)[1] = *(const uint4*)(pp + 16);
    }
#pragma unroll
    for (int nj = 0; nj < 8; ++nj) {
      v16bf vb;
      const u16* vp = &Vs[cur][(nj * 16 + ml) * 40 + half * 8];
      ((uint4*)&vb)[0] = *(const uint4*)vp;
      ((uint4*)&vb)[1] = *(const uint4*)(vp + 16);
      O[nj] = wmma_bf16(pa, vb, O[nj]);
    }
  }

  // --- finalize: O /= l, write bf16 to [B, S, D] ---
#pragma unroll
  for (int r = 0; r < 8; ++r) {
    const float inv = 1.0f / li[r];
    const int qi = qbase_w + r + 8 * half;
    u16* orow = obf + ((size_t)b * SS + qi) * DD + h * HDD;
#pragma unroll
    for (int nj = 0; nj < 8; ++nj)
      orow[nj * 16 + ml] = f2bf(O[nj][r] * inv);
  }
}

// ---------------------------------------------------------------------------
// Host side
// ---------------------------------------------------------------------------
extern "C" void kernel_launch(void* const* d_in, const int* in_sizes, int n_in,
                              void* d_out, int out_size, void* d_ws,
                              size_t ws_size, hipStream_t stream) {
  const float* x    = (const float*)d_in[0];  // [B,S,D]
  const float* Wqkv = (const float*)d_in[1];  // [D,3D]
  const float* Wout = (const float*)d_in[2];  // [D,D]
  const float* bout = (const float*)d_in[3];  // [D]
  float* out = (float*)d_out;                 // [B,S,D]

  constexpr size_t NX    = (size_t)BB * SS * DD;       // 16,777,216
  constexpr size_t NWQKV = (size_t)DD * 3 * DD;        // 12,582,912
  constexpr size_t NWOUT = (size_t)DD * DD;            //  4,194,304
  constexpr size_t NHEAD = (size_t)BB * HH * SS * HDD; // 16,777,216

  u16* p = (u16*)d_ws;
  u16* xbf    = p; p += NX;
  u16* wqkvt  = p; p += NWQKV;   // [3D][D]
  u16* woutt  = p; p += NWOUT;   // [D][D] transposed
  u16* qbf    = p; p += NHEAD;   // [B,H,S,HD]
  u16* kbf    = p; p += NHEAD;
  u16* vbf    = p; p += NHEAD;
  u16* vtbf   = p; p += NHEAD;   // [B,H,HD,S]
  u16* obf    = p; p += NX;      // [B,S,D]

  // 1) cast x; transpose+cast weights to [N][K]
  cast_bf16_kernel<<<(int)(NX / 4 / 256), 256, 0, stream>>>(x, xbf, (int)(NX / 4));
  transpose_bf16_kernel<float><<<dim3(3 * DD / 64, DD / 64, 1), 256, 0, stream>>>(
      Wqkv, wqkvt, DD, 3 * DD, 0, 0);
  transpose_bf16_kernel<float><<<dim3(DD / 64, DD / 64, 1), 256, 0, stream>>>(
      Wout, woutt, DD, DD, 0, 0);

  // 2) QKV projection: [8192,2048] x [2048,6144] -> scatter q/k/v heads
  gemm_bf16_kernel<1><<<dim3(6144 / 128, 8192 / 64), 256, 0, stream>>>(
      xbf, wqkvt, nullptr, nullptr, qbf, kbf, vbf, 8192, 6144, 2048);

  // 3) RoPE in place on q, k
  rope_kernel<<<(BB * HH * SS * (HDD / 2)) / 256, 256, 0, stream>>>(qbf, kbf);

  // 4) V -> [B,H,HD,S] for contiguous key-tile staging
  transpose_bf16_kernel<u16><<<dim3(HDD / 64, SS / 64, BB * HH), 256, 0, stream>>>(
      vbf, vtbf, SS, HDD, (size_t)SS * HDD, (size_t)SS * HDD);

  // 5) causal flash attention -> obf [B,S,D]
  attn_kernel<<<BB * HH * (SS / 64), 128, 0, stream>>>(qbf, kbf, vtbf, obf);

  // 6) output projection: [8192,2048] x [2048,2048] + bias -> fp32 out
  gemm_bf16_kernel<0><<<dim3(2048 / 128, 8192 / 64), 256, 0, stream>>>(
      obf, woutt, bout, out, nullptr, nullptr, nullptr, 8192, 2048, 2048);
}
